// EncoderRNN_22591527977138
// MI455X (gfx1250) — matrix operand order repaired and checked by reading, not compiled, not run on hardware
//
#include <hip/hip_runtime.h>

typedef __attribute__((ext_vector_type(16))) _Float16 v16h;
typedef __attribute__((ext_vector_type(8)))  _Float16 v8h;
typedef __attribute__((ext_vector_type(8)))  float    v8f;

namespace {
constexpr int S  = 2048;          // sequence length
constexpr int H  = 512;           // hidden size
constexpr int K  = 1024;          // 2H (comb = [x ; h])
constexpr int NF = 2048;          // 4 gates * H fused output dim
constexpr int KB = 32;            // K / 32 k-blocks per WMMA chain
constexpr int WGS_PER_DIR  = 16;  // workgroups per direction
constexpr int UNITS_PER_WG = H / WGS_PER_DIR;  // 32 hidden units
constexpr int TILES_PER_WG = 8;   // 4 gates * 2 n-tiles of 16
constexpr size_t FRAG_HALVES = 32 * 16;           // one (tile,kb) B fragment
constexpr size_t TILE_HALVES = KB * FRAG_HALVES;  // 16384 halves = 32 KB
// workspace layout (bytes)
constexpr size_t CNT_OFF  = 0;                       // [2][S] u32 barrier counters
constexpr size_t HBUF_OFF = (size_t)2 * S * 4;       // [2 dir][2 buf][H] f16
constexpr size_t X16_OFF  = HBUF_OFF + 2 * 2 * H * 2;
constexpr size_t BPK_OFF  = X16_OFF + (size_t)S * H * 2;
constexpr size_t SMEM_BYTES =
    TILES_PER_WG * TILE_HALVES * 2 /*weights*/ + K * 2 /*comb*/ + 128 * 4 /*gates*/;
}

__global__ void zero_f32_kernel(float* p, int n) {
  for (int i = blockIdx.x * blockDim.x + threadIdx.x; i < n; i += gridDim.x * blockDim.x)
    p[i] = 0.f;
}
__global__ void zero_u32_kernel(unsigned* p, int n) {
  for (int i = blockIdx.x * blockDim.x + threadIdx.x; i < n; i += gridDim.x * blockDim.x)
    p[i] = 0u;
}

// Embedding gather, fp32 -> f16, one block per timestep.
__global__ void embed_kernel(const int* __restrict__ ids, const float* __restrict__ emb,
                             _Float16* __restrict__ x16) {
  const int t = blockIdx.x;
  const int id = ids[t];
  for (int j = threadIdx.x; j < H; j += blockDim.x)
    x16[(size_t)t * H + j] = (_Float16)emb[(size_t)id * H + j];
}

// Pack fused-gate weights into per-lane WMMA B fragments (f16).
// packed index p = ((((dir*128 + ntile)*32 + kb)*32) + lane)*16 + e
// B 32x16 (KxN) layout: N = lane%16; K = kb*32 + e + (lane>=16 ? 16 : 0)
__global__ void pack_kernel(const float* __restrict__ Wif, const float* __restrict__ Wff,
                            const float* __restrict__ Wcf, const float* __restrict__ Wof,
                            const float* __restrict__ Wib, const float* __restrict__ Wfb,
                            const float* __restrict__ Wcb, const float* __restrict__ Wob,
                            _Float16* __restrict__ bpack) {
  const float* Wt[8] = {Wif, Wff, Wcf, Wof, Wib, Wfb, Wcb, Wob};
  unsigned p = blockIdx.x * blockDim.x + threadIdx.x;
  if (p >= (1u << 22)) return;
  const unsigned e     = p & 15u;
  const unsigned lane  = (p >> 4) & 31u;
  const unsigned kb    = (p >> 9) & 31u;
  const unsigned ntile = (p >> 14) & 127u;
  const unsigned d     = p >> 21;
  const unsigned nG    = ntile * 16 + (lane & 15);       // fused output index
  const unsigned k     = kb * 32 + e + ((lane >= 16) ? 16u : 0u);
  const unsigned gate  = nG >> 9;                        // 0..3 = i,f,c,o
  const unsigned n     = nG & 511u;
  bpack[p] = (_Float16)Wt[d * 4 + gate][(size_t)n * K + k];
}

// Persistent bidirectional LSTM. 32 blocks: blocks 0..15 = fwd, 16..31 = bwd.
// Each WG keeps its 256 KB weight slice resident in LDS for all 2048 steps.
__global__ __launch_bounds__(256, 1)
void bilstm_kernel(const _Float16* __restrict__ x16,
                   const _Float16* __restrict__ bpack,
                   _Float16* __restrict__ hbuf,   // [2 dir][2 buf][H]
                   unsigned* __restrict__ cnt,    // [2 dir][S]
                   const float* __restrict__ bi_f, const float* __restrict__ bf_f,
                   const float* __restrict__ bc_f, const float* __restrict__ bo_f,
                   const float* __restrict__ bi_b, const float* __restrict__ bf_b,
                   const float* __restrict__ bc_b, const float* __restrict__ bo_b,
                   float* __restrict__ out) {
  extern __shared__ char smem[];
  _Float16* lds_w  = (_Float16*)smem;                        // 8 * 16384 halves
  _Float16* comb   = lds_w + TILES_PER_WG * TILE_HALVES;     // 1024 halves
  float*    gatesL = (float*)(comb + K);                     // 128 floats

  const int dir  = blockIdx.x / WGS_PER_DIR;  // 0 fwd, 1 bwd
  const int r    = blockIdx.x % WGS_PER_DIR;  // slice id within direction
  const int tid  = threadIdx.x;
  const int lane = tid & 31;
  const int wave = tid >> 5;                  // 0..7

  const _Float16* bp = bpack + (size_t)dir * NF * K;
  _Float16* hb = hbuf + (size_t)dir * 2 * H;
  unsigned* mycnt = cnt + (size_t)dir * S;

  // Hoist this thread's gate biases (step-invariant).
  const int jg = r * UNITS_PER_WG + (tid < UNITS_PER_WG ? tid : 0);
  const float bi_r = (dir ? bi_b : bi_f)[jg];
  const float bf_r = (dir ? bf_b : bf_f)[jg];
  const float bc_r = (dir ? bc_b : bc_f)[jg];
  const float bo_r = (dir ? bo_b : bo_f)[jg];

  // Stage this WG's 8 weight tiles (256 KB f16) into LDS once.
  for (int lt = 0; lt < TILES_PER_WG; ++lt) {
    const int g = lt >> 1;
    const int ntile = g * (H / 16) + r * 2 + (lt & 1);
    const _Float16* src = bp + (size_t)ntile * TILE_HALVES;
    _Float16* dst = lds_w + (size_t)lt * TILE_HALVES;
    for (int j = tid * 8; j < (int)TILE_HALVES; j += 256 * 8)
      *(v8h*)(dst + j) = *(const v8h*)(src + j);
  }
  __syncthreads();

  float c_reg = 0.f, h_reg = 0.f;

  for (int s = 0; s < S; ++s) {
    const int t = dir ? (S - 1 - s) : s;

    // Phase A: comb = [x_t ; h_{prev}] in f16.
    for (int j = tid; j < K; j += 256)
      comb[j] = (j < H) ? x16[(size_t)t * H + j] : hb[(s & 1) * H + (j - H)];
    __syncthreads();

    // Phase B: each wave computes one 16-wide output tile, K=1024 via 32 WMMA.
    // A is a *broadcast* matrix (all 16 rows = comb vector; D row 0 unchanged),
    // so LDS loads are lane-uniform and EXEC stays all-ones.
    // Manual software pipeline: fragment kb+1 is loaded into a distinct
    // register pair *before* the WMMA for kb, so each WMMA only waits for
    // old loads while 4 new ds_loads stay in flight. Even/odd accumulators
    // keep two independent XDL chains per wave. Full unroll folds all DS
    // addresses into immediate offsets (kb*1024 <= 31KB fits DS offset16).
    {
      v8f acc0 = {}, acc1 = {};
      const _Float16* wt = lds_w + (size_t)wave * TILE_HALVES + (size_t)lane * 16;
      const _Float16* ca = comb + ((lane >> 4) << 4);  // 0 or 16

      auto loadA = [&](int kb) -> v16h {
        const _Float16* cp = ca + kb * 32;
        v8h lo = *(const v8h*)cp;        // K = off .. off+7
        v8h hi = *(const v8h*)(cp + 8);  // K = off+8 .. off+15
        return __builtin_shufflevector(lo, hi, 0, 1, 2, 3, 4, 5, 6, 7,
                                       8, 9, 10, 11, 12, 13, 14, 15);
      };
      auto loadB = [&](int kb) -> v16h {
        return *(const v16h*)(wt + (size_t)kb * FRAG_HALVES);  // 2x ds_load_b128
      };

      v16h a_cur = loadA(0);
      v16h b_cur = loadB(0);
#pragma unroll
      for (int kb = 0; kb < KB; ++kb) {
        v16h a_nxt = a_cur, b_nxt = b_cur;
        if (kb + 1 < KB) {               // static under full unroll
          a_nxt = loadA(kb + 1);
          b_nxt = loadB(kb + 1);
        }
        if (kb & 1)
          acc1 = __builtin_amdgcn_wmma_f32_16x16x32_f16(
              false, a_cur, false, b_cur, (short)0, acc1, false, false);
        else
          acc0 = __builtin_amdgcn_wmma_f32_16x16x32_f16(
              false, a_cur, false, b_cur, (short)0, acc0, false, false);
        a_cur = a_nxt;
        b_cur = b_nxt;
      }
      if (lane < 16) {  // D row M=0 lives in VGPR0 of lanes 0..15
        const int g = wave >> 1;
        gatesL[g * 32 + (wave & 1) * 16 + lane] = acc0[0] + acc1[0];
      }
    }
    __syncthreads();

    // Phase C: fused state update for this WG's 32 units; publish h slice.
    if (tid < UNITS_PER_WG) {
      const float gi = 1.f / (1.f + __expf(-(gatesL[tid]      + bi_r)));
      const float gf = 1.f / (1.f + __expf(-(gatesL[32 + tid] + bf_r)));
      const float gc = tanhf(gatesL[64 + tid] + bc_r);
      const float go = 1.f / (1.f + __expf(-(gatesL[96 + tid] + bo_r)));
      c_reg = gf * c_reg + gi * gc;
      h_reg = go * tanhf(c_reg);
      hb[((s + 1) & 1) * H + jg] = (_Float16)h_reg;
      if (dir) out[(size_t)t * (2 * H) + H + jg] = h_reg;  // bwd hidden per step
    }

    // Inter-WG barrier for this direction (single-use counter per step).
    __threadfence();
    __syncthreads();
    if (tid == 0) {
      atomicAdd(&mycnt[s], 1u);
      while (atomicAdd(&mycnt[s], 0u) < (unsigned)WGS_PER_DIR)
        __builtin_amdgcn_s_sleep(1);
    }
    __syncthreads();
    __threadfence();
  }

  // Epilogue: fwd hidden only at last row; final_hidden = [h_fwd ; h_bwd].
  if (tid < UNITS_PER_WG) {
    if (dir == 0) out[(size_t)(S - 1) * (2 * H) + jg] = h_reg;
    out[(size_t)S * (2 * H) + dir * H + jg] = h_reg;
  }
}

extern "C" void kernel_launch(void* const* d_in, const int* in_sizes, int n_in,
                              void* d_out, int out_size, void* d_ws, size_t ws_size,
                              hipStream_t stream) {
  (void)in_sizes; (void)n_in; (void)out_size; (void)ws_size;
  const int*   ids = (const int*)d_in[0];
  const float* emb = (const float*)d_in[2];
  const float* W[8];
  const float* B[8];
  for (int g = 0; g < 8; ++g) {
    W[g] = (const float*)d_in[3 + 2 * g];
    B[g] = (const float*)d_in[4 + 2 * g];
  }
  float* out = (float*)d_out;
  char* ws = (char*)d_ws;
  unsigned* cnt   = (unsigned*)(ws + CNT_OFF);
  _Float16* hbuf  = (_Float16*)(ws + HBUF_OFF);
  _Float16* x16   = (_Float16*)(ws + X16_OFF);
  _Float16* bpack = (_Float16*)(ws + BPK_OFF);

  // Allow 258 KB dynamic LDS (CDNA5 WGP has 320 KB). Deterministic, capture-safe.
  hipFuncSetAttribute(reinterpret_cast<const void*>(bilstm_kernel),
                      hipFuncAttributeMaxDynamicSharedMemorySize, (int)SMEM_BYTES);

  zero_f32_kernel<<<256, 256, 0, stream>>>(out, S * 2 * H + 2 * H);
  zero_u32_kernel<<<16, 256, 0, stream>>>(cnt, (int)((HBUF_OFF + 2 * 2 * H * 2) / 4));
  embed_kernel<<<S, 256, 0, stream>>>(ids, emb, x16);
  pack_kernel<<<(1 << 22) / 256, 256, 0, stream>>>(W[0], W[1], W[2], W[3],
                                                   W[4], W[5], W[6], W[7], bpack);
  bilstm_kernel<<<dim3(2 * WGS_PER_DIR), dim3(256), SMEM_BYTES, stream>>>(
      x16, bpack, hbuf, cnt,
      B[0], B[1], B[2], B[3], B[4], B[5], B[6], B[7], out);
}
